// DRNNCell_75196287418408
// MI455X (gfx1250) — matrix operand order repaired and checked by reading, not compiled, not run on hardware
//
#include <hip/hip_runtime.h>
#include <hip/hip_bf16.h>
#include <cstdint>
#include <cstddef>

// ---------------------------------------------------------------------------
// Problem constants (from reference)
// ---------------------------------------------------------------------------
constexpr int Bsz = 1024, Ssz = 2, Tsz = 200;
constexpr int Fd = 100, Gd = 500, Pd = 500, Od = 300;

// Padded GEMM dims (multiples of 32 so WMMA tiles have no tails)
constexpr int KP_G_IH = 608;   // F+P = 600
constexpr int KP_G_HH = 512;   // G   = 500
constexpr int NP_G    = 1504;  // 3G  = 1500
constexpr int KP_P_IH = 608;   // F+G = 600
constexpr int KP_P_HH = 512;   // P   = 500
constexpr int NP_P    = 1504;  // 3P  = 1500
constexpr int KP_O_IH = 512;   // P   = 500
constexpr int KP_O_HH = 320;   // O   = 300
constexpr int NP_O    = 928;   // 3O  = 900

typedef __attribute__((ext_vector_type(16))) __bf16 v16bf;
typedef __attribute__((ext_vector_type(8)))  __bf16 v8bf;
typedef __attribute__((ext_vector_type(8)))  float  v8f;

__device__ __forceinline__ float sigm(float x) { return 1.0f / (1.0f + expf(-x)); }

// ---------------------------------------------------------------------------
// Attention: scores[b,t] = dot(global_history[t,b,:], w_att)
// one block per b, 8 waves, wave handles a t, lanes reduce over G
// ---------------------------------------------------------------------------
__global__ void k_scores(const float* __restrict__ gh, const float* __restrict__ watt,
                         float* __restrict__ scores) {
  __shared__ float satt[Gd];
  const int b = blockIdx.x;
  for (int d = threadIdx.x; d < Gd; d += blockDim.x) satt[d] = watt[d];
  __syncthreads();
  const int wave = threadIdx.x >> 5;
  const int lane = threadIdx.x & 31;
  for (int t = wave; t < Tsz; t += 8) {
    const float* row = gh + ((size_t)t * Bsz + b) * Gd;
    float p = 0.f;
    for (int d = lane; d < Gd; d += 32) p += row[d] * satt[d];
#pragma unroll
    for (int off = 16; off > 0; off >>= 1) p += __shfl_xor(p, off, 32);
    if (lane == 0) scores[(size_t)b * Tsz + t] = p;
  }
}

// softmax over t (T=200 <= 256 threads); writes alpha scratch + alpha output
__global__ void k_softmax(const float* __restrict__ scores, float* __restrict__ alpha,
                          float* __restrict__ alpha_out) {
  __shared__ float red[256];
  const int b = blockIdx.x, tid = threadIdx.x;
  float v = (tid < Tsz) ? scores[(size_t)b * Tsz + tid] : -3.4e38f;
  red[tid] = v;
  __syncthreads();
  for (int s = 128; s > 0; s >>= 1) {
    if (tid < s) red[tid] = fmaxf(red[tid], red[tid + s]);
    __syncthreads();
  }
  const float mx = red[0];
  __syncthreads();
  float e = (tid < Tsz) ? expf(v - mx) : 0.f;
  red[tid] = e;
  __syncthreads();
  for (int s = 128; s > 0; s >>= 1) {
    if (tid < s) red[tid] += red[tid + s];
    __syncthreads();
  }
  const float inv = 1.0f / red[0];
  if (tid < Tsz) {
    const float a = e * inv;
    alpha[(size_t)b * Tsz + tid]     = a;
    alpha_out[(size_t)b * Tsz + tid] = a;
  }
}

// context[b,d] = sum_t alpha[b,t] * gh[t,b,d]   (bandwidth-bound streaming pass)
__global__ void k_context(const float* __restrict__ alpha, const float* __restrict__ gh,
                          float* __restrict__ ctx) {
  __shared__ float sa[Tsz];
  const int b = blockIdx.x, d = threadIdx.x;
  for (int t = threadIdx.x; t < Tsz; t += blockDim.x) sa[t] = alpha[(size_t)b * Tsz + t];
  __syncthreads();
  if (d >= Gd) return;
  float acc = 0.f;
  for (int t = 0; t < Tsz; ++t) {
    const float* p = gh + ((size_t)t * Bsz + b) * Gd + d;
    if (t + 1 < Tsz) __builtin_prefetch(gh + ((size_t)(t + 1) * Bsz + b) * Gd + d, 0, 1);
    acc += sa[t] * (*p);
  }
  ctx[(size_t)b * Gd + d] = acc;
}

// ---------------------------------------------------------------------------
// Packing kernels: f32 -> zero-padded bf16 matrices for WMMA
// ---------------------------------------------------------------------------
__global__ void k_pack_weight(const float* __restrict__ W, __bf16* __restrict__ out,
                              int N, int K, int Np, int Kp) {
  const size_t total = (size_t)Np * Kp;
  for (size_t i = (size_t)blockIdx.x * blockDim.x + threadIdx.x; i < total;
       i += (size_t)gridDim.x * blockDim.x) {
    const int n = (int)(i / Kp), k = (int)(i % Kp);
    float v = (n < N && k < K) ? W[(size_t)n * K + k] : 0.f;
    out[i] = (__bf16)v;
  }
}

// generic: dense f32 rows -> padded bf16 rows
__global__ void k_pack_rows(const float* __restrict__ src, long srcStride,
                            __bf16* __restrict__ dst, int rows, int D, int Kp) {
  const size_t total = (size_t)rows * Kp;
  for (size_t i = (size_t)blockIdx.x * blockDim.x + threadIdx.x; i < total;
       i += (size_t)gridDim.x * blockDim.x) {
    const int r = (int)(i / Kp), c = (int)(i % Kp);
    float v = (c < D) ? src[(size_t)r * srcStride + c] : 0.f;
    dst[i] = (__bf16)v;
  }
}

// Xg = [input | initial_party[b, argmax(mask)] ] padded to 608
__global__ void k_pack_xg(const float* __restrict__ input, const float* __restrict__ pmask,
                          const float* __restrict__ iparty, __bf16* __restrict__ dst) {
  const size_t total = (size_t)Bsz * KP_G_IH;
  for (size_t i = (size_t)blockIdx.x * blockDim.x + threadIdx.x; i < total;
       i += (size_t)gridDim.x * blockDim.x) {
    const int b = (int)(i / KP_G_IH), c = (int)(i % KP_G_IH);
    float v = 0.f;
    if (c < Fd) v = input[(size_t)b * Fd + c];
    else if (c < Fd + Pd) {
      const int id = (pmask[(size_t)b * Ssz + 1] > pmask[(size_t)b * Ssz]) ? 1 : 0;
      v = iparty[((size_t)b * Ssz + id) * Pd + (c - Fd)];
    }
    dst[i] = (__bf16)v;
  }
}

// Xp rows (b*2+s) = [input_b | context_b] padded to 608 (same for both s)
__global__ void k_pack_xp(const float* __restrict__ input, const float* __restrict__ ctx,
                          __bf16* __restrict__ dst) {
  const size_t total = (size_t)Bsz * Ssz * KP_P_IH;
  for (size_t i = (size_t)blockIdx.x * blockDim.x + threadIdx.x; i < total;
       i += (size_t)gridDim.x * blockDim.x) {
    const int row = (int)(i / KP_P_IH), c = (int)(i % KP_P_IH);
    const int b = row >> 1;
    float v = 0.f;
    if (c < Fd) v = input[(size_t)b * Fd + c];
    else if (c < Fd + Gd) v = ctx[(size_t)b * Gd + (c - Fd)];
    dst[i] = (__bf16)v;
  }
}

// Xo = party_new[b, argmax(mask)] gathered from d_out region, padded to 512
__global__ void k_pack_xo(const float* __restrict__ party_new, const float* __restrict__ pmask,
                          __bf16* __restrict__ dst) {
  const size_t total = (size_t)Bsz * KP_O_IH;
  for (size_t i = (size_t)blockIdx.x * blockDim.x + threadIdx.x; i < total;
       i += (size_t)gridDim.x * blockDim.x) {
    const int b = (int)(i / KP_O_IH), c = (int)(i % KP_O_IH);
    float v = 0.f;
    if (c < Pd) {
      const int id = (pmask[(size_t)b * Ssz + 1] > pmask[(size_t)b * Ssz]) ? 1 : 0;
      v = party_new[((size_t)b * Ssz + id) * Pd + c];
    }
    dst[i] = (__bf16)v;
  }
}

// ---------------------------------------------------------------------------
// WMMA NT-GEMM:  C[M,Np] = A[M,Kp](bf16) * W[Np,Kp]^T(bf16) + bias  (f32 acc)
// 8 waves/block; each wave owns a 32x32 C tile = 2x2 v_wmma_f32_16x16x32_bf16
// A/B fragments use contiguous k: lane half h holds k 16h..16h+15 in both (16x32 / 32x16).
// ---------------------------------------------------------------------------
__global__ void k_gemm_bf16(const __bf16* __restrict__ A, const __bf16* __restrict__ W,
                            const float* __restrict__ bias, float* __restrict__ C,
                            int M, int Np, int Kp, int Nreal) {
  const int wave = threadIdx.x >> 5;
  const int lane = threadIdx.x & 31;
  const int half = lane >> 4;   // 0: lanes 0-15, 1: lanes 16-31
  const int r    = lane & 15;
  const int nTN = Np >> 5;
  const int nTM = M >> 5;
  const int tile = blockIdx.x * 8 + wave;
  if (tile >= nTM * nTN) return;
  const int m0 = (tile / nTN) << 5;
  const int n0 = (tile % nTN) << 5;

  v8f acc[2][2] = {};
  const __bf16* Arow0 = A + (size_t)(m0 + r)      * Kp;
  const __bf16* Arow1 = A + (size_t)(m0 + 16 + r) * Kp;
  const __bf16* Wrow0 = W + (size_t)(n0 + r)      * Kp + half * 16;
  const __bf16* Wrow1 = W + (size_t)(n0 + 16 + r) * Kp + half * 16;

  for (int k0 = 0; k0 < Kp; k0 += 32) {
    // A fragment: lanes 0-15 hold row m, K {0..15}; lanes 16-31 hold K {16..31}
    v8bf a0lo = *(const v8bf*)(Arow0 + k0 + half * 16);
    v8bf a0hi = *(const v8bf*)(Arow0 + k0 + half * 16 + 8);
    v8bf a1lo = *(const v8bf*)(Arow1 + k0 + half * 16);
    v8bf a1hi = *(const v8bf*)(Arow1 + k0 + half * 16 + 8);
    v16bf a0, a1;
#pragma unroll
    for (int i = 0; i < 8; ++i) {
      a0[i] = a0lo[i]; a0[i + 8] = a0hi[i];
      a1[i] = a1lo[i]; a1[i + 8] = a1hi[i];
    }
    // B fragment: lane n holds W[n0+r][k0 + half*16 .. +15] (B[K][N] = W[N][K])
    v16bf b0 = *(const v16bf*)(Wrow0 + k0);
    v16bf b1 = *(const v16bf*)(Wrow1 + k0);

    acc[0][0] = __builtin_amdgcn_wmma_f32_16x16x32_bf16(false, a0, false, b0, (short)0, acc[0][0], false, false);
    acc[0][1] = __builtin_amdgcn_wmma_f32_16x16x32_bf16(false, a0, false, b1, (short)0, acc[0][1], false, false);
    acc[1][0] = __builtin_amdgcn_wmma_f32_16x16x32_bf16(false, a1, false, b0, (short)0, acc[1][0], false, false);
    acc[1][1] = __builtin_amdgcn_wmma_f32_16x16x32_bf16(false, a1, false, b1, (short)0, acc[1][1], false, false);
  }

  // C/D layout: acc[i], lanes 0-15 -> M=i, N=lane; lanes 16-31 -> M=8+i, N=lane-16
#pragma unroll
  for (int mi = 0; mi < 2; ++mi) {
#pragma unroll
    for (int ni = 0; ni < 2; ++ni) {
      const int col = n0 + ni * 16 + r;
      const float bv = (col < Nreal) ? bias[col] : 0.f;
#pragma unroll
      for (int i = 0; i < 8; ++i) {
        const int row = m0 + mi * 16 + half * 8 + i;
        C[(size_t)row * Np + col] = acc[mi][ni][i] + bv;
      }
    }
  }
}

// ---------------------------------------------------------------------------
// GRU elementwise epilogue (optionally blended with party mask)
// ---------------------------------------------------------------------------
__global__ void k_gru(const float* __restrict__ gi, const float* __restrict__ gh,
                      int Np, int H, int rows,
                      const float* __restrict__ hprev, long hstride,
                      const float* __restrict__ mask, int useMask,
                      float* __restrict__ out) {
  const size_t total = (size_t)rows * H;
  for (size_t i = (size_t)blockIdx.x * blockDim.x + threadIdx.x; i < total;
       i += (size_t)gridDim.x * blockDim.x) {
    const int rw = (int)(i / H), j = (int)(i % H);
    const size_t base = (size_t)rw * Np;
    const float ir = gi[base + j], iz = gi[base + H + j], in = gi[base + 2 * H + j];
    const float hr = gh[base + j], hz = gh[base + H + j], hn = gh[base + 2 * H + j];
    const float rg = sigm(ir + hr);
    const float zg = sigm(iz + hz);
    const float ng = tanhf(in + rg * hn);
    const float hp = hprev[(size_t)rw * hstride + j];
    float nh = (1.f - zg) * ng + zg * hp;
    if (useMask) {
      const float m = mask[rw];
      nh = hp * (1.f - m) + nh * m;
    }
    out[(size_t)rw * H + j] = nh;
  }
}

// ---------------------------------------------------------------------------
// Host launcher
// ---------------------------------------------------------------------------
static inline char* bump(char*& p, size_t bytes) {
  char* r = p;
  p += (bytes + 255) & ~(size_t)255;
  return r;
}

extern "C" void kernel_launch(void* const* d_in, const int* in_sizes, int n_in,
                              void* d_out, int out_size, void* d_ws, size_t ws_size,
                              hipStream_t stream) {
  (void)in_sizes; (void)n_in; (void)out_size; (void)ws_size;
  const float* input    = (const float*)d_in[0];
  const float* pmask    = (const float*)d_in[1];
  const float* ghist    = (const float*)d_in[2];
  const float* iparty   = (const float*)d_in[3];
  const float* ioutput  = (const float*)d_in[4];
  const float* w_ih_g   = (const float*)d_in[5];
  const float* w_hh_g   = (const float*)d_in[6];
  const float* b_ih_g   = (const float*)d_in[7];
  const float* b_hh_g   = (const float*)d_in[8];
  const float* w_ih_p   = (const float*)d_in[9];
  const float* w_hh_p   = (const float*)d_in[10];
  const float* b_ih_p   = (const float*)d_in[11];
  const float* b_hh_p   = (const float*)d_in[12];
  const float* w_ih_o   = (const float*)d_in[13];
  const float* w_hh_o   = (const float*)d_in[14];
  const float* b_ih_o   = (const float*)d_in[15];
  const float* b_hh_o   = (const float*)d_in[16];
  const float* w_att    = (const float*)d_in[17];

  float* out = (float*)d_out;
  float* out_gstate = out;                                   // 1024*500
  float* out_party  = out + (size_t)Bsz * Gd;                // 1024*2*500
  float* out_output = out_party + (size_t)Bsz * Ssz * Pd;    // 1024*300
  float* out_alpha  = out_output + (size_t)Bsz * Od;         // 1024*200

  // workspace carve-up (buffers reused across sequential GRU stages)
  char* p = (char*)d_ws;
  float*  scores = (float*)bump(p, (size_t)Bsz * Tsz * 4);
  float*  alpha  = (float*)bump(p, (size_t)Bsz * Tsz * 4);
  float*  ctx    = (float*)bump(p, (size_t)Bsz * Gd * 4);
  __bf16* Xb     = (__bf16*)bump(p, (size_t)Bsz * Ssz * KP_P_IH * 2);  // max X
  __bf16* Hb     = (__bf16*)bump(p, (size_t)Bsz * Ssz * KP_P_HH * 2);  // max H
  __bf16* Wih    = (__bf16*)bump(p, (size_t)NP_G * KP_G_IH * 2);       // max W_ih
  __bf16* Whh    = (__bf16*)bump(p, (size_t)NP_G * KP_G_HH * 2);       // max W_hh
  float*  GI     = (float*)bump(p, (size_t)Bsz * Ssz * NP_P * 4);      // max gi
  float*  GH     = (float*)bump(p, (size_t)Bsz * Ssz * NP_P * 4);      // max gh

  const int TPB = 256;
  auto gblk = [](size_t n, int tpb) { return (int)((n + tpb - 1) / tpb); };

  // ---- attention (streams global_history twice; dominant HBM traffic) ----
  k_scores <<<Bsz, 256, 0, stream>>>(ghist, w_att, scores);
  k_softmax<<<Bsz, 256, 0, stream>>>(scores, alpha, out_alpha);
  k_context<<<Bsz, 512, 0, stream>>>(alpha, ghist, ctx);

  // ---- stage G: global GRU ----
  k_pack_weight<<<gblk((size_t)NP_G * KP_G_IH, TPB), TPB, 0, stream>>>(w_ih_g, Wih, 3 * Gd, Fd + Pd, NP_G, KP_G_IH);
  k_pack_weight<<<gblk((size_t)NP_G * KP_G_HH, TPB), TPB, 0, stream>>>(w_hh_g, Whh, 3 * Gd, Gd, NP_G, KP_G_HH);
  k_pack_xg   <<<gblk((size_t)Bsz * KP_G_IH, TPB), TPB, 0, stream>>>(input, pmask, iparty, Xb);
  const float* h_last = ghist + (size_t)(Tsz - 1) * Bsz * Gd;
  k_pack_rows <<<gblk((size_t)Bsz * KP_G_HH, TPB), TPB, 0, stream>>>(h_last, Gd, Hb, Bsz, Gd, KP_G_HH);
  k_gemm_bf16<<<((Bsz / 32) * (NP_G / 32) + 7) / 8, 256, 0, stream>>>(Xb, Wih, b_ih_g, GI, Bsz, NP_G, KP_G_IH, 3 * Gd);
  k_gemm_bf16<<<((Bsz / 32) * (NP_G / 32) + 7) / 8, 256, 0, stream>>>(Hb, Whh, b_hh_g, GH, Bsz, NP_G, KP_G_HH, 3 * Gd);
  k_gru<<<gblk((size_t)Bsz * Gd, TPB), TPB, 0, stream>>>(GI, GH, NP_G, Gd, Bsz, h_last, Gd, nullptr, 0, out_gstate);

  // ---- stage P: party GRU (rows = B*S = 2048) ----
  k_pack_weight<<<gblk((size_t)NP_P * KP_P_IH, TPB), TPB, 0, stream>>>(w_ih_p, Wih, 3 * Pd, Fd + Gd, NP_P, KP_P_IH);
  k_pack_weight<<<gblk((size_t)NP_P * KP_P_HH, TPB), TPB, 0, stream>>>(w_hh_p, Whh, 3 * Pd, Pd, NP_P, KP_P_HH);
  k_pack_xp   <<<gblk((size_t)Bsz * Ssz * KP_P_IH, TPB), TPB, 0, stream>>>(input, ctx, Xb);
  k_pack_rows <<<gblk((size_t)Bsz * Ssz * KP_P_HH, TPB), TPB, 0, stream>>>(iparty, Pd, Hb, Bsz * Ssz, Pd, KP_P_HH);
  k_gemm_bf16<<<((Bsz * Ssz / 32) * (NP_P / 32) + 7) / 8, 256, 0, stream>>>(Xb, Wih, b_ih_p, GI, Bsz * Ssz, NP_P, KP_P_IH, 3 * Pd);
  k_gemm_bf16<<<((Bsz * Ssz / 32) * (NP_P / 32) + 7) / 8, 256, 0, stream>>>(Hb, Whh, b_hh_p, GH, Bsz * Ssz, NP_P, KP_P_HH, 3 * Pd);
  k_gru<<<gblk((size_t)Bsz * Ssz * Pd, TPB), TPB, 0, stream>>>(GI, GH, NP_P, Pd, Bsz * Ssz, iparty, Pd, pmask, 1, out_party);

  // ---- stage O: output GRU ----
  k_pack_weight<<<gblk((size_t)NP_O * KP_O_IH, TPB), TPB, 0, stream>>>(w_ih_o, Wih, 3 * Od, Pd, NP_O, KP_O_IH);
  k_pack_weight<<<gblk((size_t)NP_O * KP_O_HH, TPB), TPB, 0, stream>>>(w_hh_o, Whh, 3 * Od, Od, NP_O, KP_O_HH);
  k_pack_xo   <<<gblk((size_t)Bsz * KP_O_IH, TPB), TPB, 0, stream>>>(out_party, pmask, Xb);
  k_pack_rows <<<gblk((size_t)Bsz * KP_O_HH, TPB), TPB, 0, stream>>>(ioutput, Od, Hb, Bsz, Od, KP_O_HH);
  k_gemm_bf16<<<((Bsz / 32) * (NP_O / 32) + 7) / 8, 256, 0, stream>>>(Xb, Wih, b_ih_o, GI, Bsz, NP_O, KP_O_IH, 3 * Od);
  k_gemm_bf16<<<((Bsz / 32) * (NP_O / 32) + 7) / 8, 256, 0, stream>>>(Hb, Whh, b_hh_o, GH, Bsz, NP_O, KP_O_HH, 3 * Od);
  k_gru<<<gblk((size_t)Bsz * Od, TPB), TPB, 0, stream>>>(GI, GH, NP_O, Od, Bsz, ioutput, Od, nullptr, 0, out_output);
}
